// SelfAttentionModule_40080634806342
// MI455X (gfx1250) — matrix operand order repaired and checked by reading, not compiled, not run on hardware
//
#include <hip/hip_runtime.h>
#include <hip/hip_bf16.h>
#include <stdint.h>

// ---------------------------------------------------------------------------
// Problem constants (reference: B=2, S=2048, D=1024, H=16, HD=64)
// ---------------------------------------------------------------------------
#define BATCH 2
#define SEQ   2048
#define DMODEL 1024
#define NHEAD 16
#define HDIM  64
#define MROWS (BATCH * SEQ)      // 4096
#define ATT_SCALE 0.125f         // 1/sqrt(64)

typedef __attribute__((ext_vector_type(16))) __bf16 bf16x16;
typedef __attribute__((ext_vector_type(8)))  float  f32x8;

union FragAB {
  uint4   u[2];   // 32 bytes = 16 bf16
  bf16x16 v;
};

__device__ __forceinline__ unsigned short f2bf(float f) {
  union { float f; unsigned u; } x; x.f = f;
  unsigned r = x.u + 0x7FFFu + ((x.u >> 16) & 1u);   // round-to-nearest-even
  return (unsigned short)(r >> 16);
}

// Generic (flat) pointer to a __shared__ object: low 32 bits are the LDS
// byte offset (LDS aperture, ISA 10.2).
__device__ __forceinline__ unsigned lds_off(const void* p) {
  return (unsigned)(uintptr_t)p;
}

// Async copy 16B global -> LDS, tracked by ASYNCcnt (CDNA5 ISA).
__device__ __forceinline__ void async_copy_b128(unsigned lds_addr, const void* gptr) {
  asm volatile("global_load_async_to_lds_b128 %0, %1, off"
               :: "v"(lds_addr), "v"((unsigned long long)(uintptr_t)gptr)
               : "memory");
}
__device__ __forceinline__ void wait_async0() {
  asm volatile("s_wait_asynccnt 0x0" ::: "memory");
}

// ---------------------------------------------------------------------------
// Kernel 0a: fp32 -> bf16 row-major copy (for x)
// ---------------------------------------------------------------------------
__global__ __launch_bounds__(256)
void cvt_f32_bf16(const float* __restrict__ in, unsigned short* __restrict__ out, int n4) {
  for (int i = blockIdx.x * blockDim.x + threadIdx.x; i < n4; i += gridDim.x * blockDim.x) {
    float4 a = reinterpret_cast<const float4*>(in)[i];
    uint2 p;
    p.x = (unsigned)f2bf(a.x) | ((unsigned)f2bf(a.y) << 16);
    p.y = (unsigned)f2bf(a.z) | ((unsigned)f2bf(a.w) << 16);
    reinterpret_cast<uint2*>(out)[i] = p;
  }
}

// ---------------------------------------------------------------------------
// Kernel 0b: weight transpose + convert: W[K,N] f32 -> WT[N,K] bf16.
// 64x64 LDS tiles; coalesced float4 reads, coalesced 8B writes.
// blockIdx.z selects which of the 4 weights.
// ---------------------------------------------------------------------------
__global__ __launch_bounds__(256)
void wtrans_bf16(const float* w0, const float* w1, const float* w2, const float* w3,
                 unsigned short* o0, unsigned short* o1,
                 unsigned short* o2, unsigned short* o3) {
  __shared__ unsigned short s[64][72];
  const float* W; unsigned short* O;
  switch (blockIdx.z) {
    case 0: W = w0; O = o0; break;
    case 1: W = w1; O = o1; break;
    case 2: W = w2; O = o2; break;
    default: W = w3; O = o3; break;
  }
  const int t = threadIdx.x;
  const int kt = blockIdx.y * 64, nt = blockIdx.x * 64;
#pragma unroll
  for (int i = 0; i < 4; ++i) {
    int row = (t >> 4) + i * 16;         // k within tile
    int col = (t & 15) * 4;              // n within tile
    float4 a = *reinterpret_cast<const float4*>(&W[(size_t)(kt + row) * DMODEL + nt + col]);
    s[row][col + 0] = f2bf(a.x); s[row][col + 1] = f2bf(a.y);
    s[row][col + 2] = f2bf(a.z); s[row][col + 3] = f2bf(a.w);
  }
  __syncthreads();
#pragma unroll
  for (int i = 0; i < 4; ++i) {
    int n  = (t >> 4) + i * 16;
    int k4 = (t & 15) * 4;
    uint2 p;
    p.x = (unsigned)s[k4 + 0][n] | ((unsigned)s[k4 + 1][n] << 16);
    p.y = (unsigned)s[k4 + 2][n] | ((unsigned)s[k4 + 3][n] << 16);
    *reinterpret_cast<uint2*>(&O[(size_t)(nt + n) * DMODEL + kt + k4]) = p;
  }
}

// ---------------------------------------------------------------------------
// Tiled bf16 WMMA GEMM:  C = A[M,K](bf16) * BT[N,K](bf16)^T
//   256 threads = 8 waves; WG tile 128x128, wave tile 32x64 (2x4 accum).
//   Double-buffered LDS fed by GLOBAL_LOAD_ASYNC_TO_LDS_B128.
//   out_mode: 1 = bf16 [b,h,s,hd] (Q/K), 2 = bf16 [b,h,hd,s] (V^T),
//             3 = f32 row-major [M,N] (final projection -> d_out)
// ---------------------------------------------------------------------------
#define GBM 128
#define GBN 128
#define GBK 32
#define LDSA 40   // padded row stride in bf16 (80B, multiple of 16B)
#define LDSB 40

__global__ __launch_bounds__(256)
void gemm_bf16_wmma(const unsigned short* __restrict__ A,
                    const unsigned short* __restrict__ BT,
                    void* __restrict__ C, int M, int N, int K, int out_mode)
{
  __shared__ unsigned short sA[2][GBM * LDSA];
  __shared__ unsigned short sB[2][GBN * LDSB];

  const int tid  = threadIdx.x;
  const int wave = tid >> 5;
  const int lane = tid & 31;
  const int lm   = lane & 15;
  const int lh   = lane >> 4;

  const int bn = blockIdx.x * GBN;
  const int bm = blockIdx.y * GBM;
  const int wm = (wave & 3) * 32;      // 4 waves along M
  const int wn = (wave >> 2) * 64;     // 2 waves along N

  f32x8 acc[2][4];
#pragma unroll
  for (int i = 0; i < 2; ++i)
#pragma unroll
    for (int j = 0; j < 4; ++j) acc[i][j] = f32x8{};

  auto stage = [&](int buf, int k0) {
    // A tile: 128 rows x 32k = 512 x 16B chunks
#pragma unroll
    for (int i = 0; i < 2; ++i) {
      int c = tid + i * 256;
      int row = c >> 2, kc = (c & 3) * 8;
      async_copy_b128(lds_off(&sA[buf][row * LDSA + kc]),
                      &A[(size_t)(bm + row) * K + k0 + kc]);
    }
    // B tile (rows of BT): 128 x 32k = 512 x 16B chunks
#pragma unroll
    for (int i = 0; i < 2; ++i) {
      int c = tid + i * 256;
      int row = c >> 2, kc = (c & 3) * 8;
      async_copy_b128(lds_off(&sB[buf][row * LDSB + kc]),
                      &BT[(size_t)(bn + row) * K + k0 + kc]);
    }
    if (k0 + GBK < K) {   // prefetch one more tile ahead into L2
      __builtin_prefetch(&A[(size_t)(bm + (tid >> 1)) * K + k0 + GBK]);
      __builtin_prefetch(&BT[(size_t)(bn + (tid >> 1)) * K + k0 + GBK]);
    }
  };

  stage(0, 0);
  wait_async0();
  __syncthreads();

  int buf = 0;
  for (int k0 = 0; k0 < K; k0 += GBK) {
    if (k0 + GBK < K) stage(buf ^ 1, k0 + GBK);

    FragAB fa[2], fb[4];
#pragma unroll
    for (int i = 0; i < 2; ++i) {
      const unsigned short* p = &sA[buf][(wm + i * 16 + lm) * LDSA + (lh ? 8 : 0)];
      fa[i].u[0] = *reinterpret_cast<const uint4*>(p);
      fa[i].u[1] = *reinterpret_cast<const uint4*>(p + 16);
    }
#pragma unroll
    for (int j = 0; j < 4; ++j) {
      const unsigned short* p = &sB[buf][(wn + j * 16 + lm) * LDSB + (lh ? 8 : 0)];
      fb[j].u[0] = *reinterpret_cast<const uint4*>(p);
      fb[j].u[1] = *reinterpret_cast<const uint4*>(p + 16);
    }
#pragma unroll
    for (int i = 0; i < 2; ++i)
#pragma unroll
      for (int j = 0; j < 4; ++j)
        acc[i][j] = __builtin_amdgcn_wmma_f32_16x16x32_bf16(
            false, fa[i].v, false, fb[j].v, (short)0, acc[i][j], false, false);

    wait_async0();
    __syncthreads();
    buf ^= 1;
  }

  // ---- epilogue: C layout = lanes hold N, VGPR r holds M=r(+8 hi half)
#pragma unroll
  for (int i = 0; i < 2; ++i) {
#pragma unroll
    for (int j = 0; j < 4; ++j) {
#pragma unroll
      for (int r = 0; r < 8; ++r) {
        int m = bm + wm + i * 16 + r + 8 * lh;
        int n = bn + wn + j * 16 + lm;
        float val = acc[i][j][r];
        if (out_mode == 3) {
          reinterpret_cast<float*>(C)[(size_t)m * N + n] = val;
        } else {
          int b = m >> 11, s = m & 2047, h = n >> 6, hd = n & 63;
          size_t idx;
          if (out_mode == 1)       // [b,h,s,hd]
            idx = (((size_t)(b * NHEAD + h) << 11) + s) * HDIM + hd;
          else                     // [b,h,hd,s]
            idx = (((size_t)(b * NHEAD + h) * HDIM + hd) << 11) + s;
          reinterpret_cast<unsigned short*>(C)[idx] = f2bf(val);
        }
      }
    }
  }
}

// ---------------------------------------------------------------------------
// Flash attention: 1 WG per (b*h, 64 q-rows). 128 threads = 4 waves.
// K/V blocks (32 keys) staged into double-buffered LDS via async copies
// (once per WG instead of once per wave). Online softmax in fp32.
//   Q,K: bf16 [b,h,s,hd]   Vt: bf16 [b,h,hd,s]   O: bf16 [b,s,d]
// ---------------------------------------------------------------------------
#define LDK 80    // 64 hd + pad (160B, 16B multiple)
#define LDV 40    // 32 keys + pad (80B, 16B multiple)

__global__ __launch_bounds__(128)
void attn_fwd_wmma(const unsigned short* __restrict__ Q,
                   const unsigned short* __restrict__ Kd,
                   const unsigned short* __restrict__ Vt,
                   unsigned short* __restrict__ O)
{
  __shared__ unsigned short sK[2][32 * LDK];   // [key][hd]
  __shared__ unsigned short sV[2][64 * LDV];   // [hd][key]
  __shared__ unsigned short sP[4][16 * 40];    // per-wave P transpose

  const int tid  = threadIdx.x;
  const int wave = tid >> 5;
  const int lane = tid & 31;
  const int lm   = lane & 15;
  const int lh   = lane >> 4;

  const int qb = blockIdx.x * 64;
  const int bh = blockIdx.y;               // b*NHEAD + h

  // Q A-fragments (2 head-dim chunks of 32), resident for whole pass
  FragAB aq[2];
  {
    int row = qb + wave * 16 + lm;
    const unsigned short* base = Q + (((size_t)bh << 11) + row) * HDIM;
#pragma unroll
    for (int c = 0; c < 2; ++c) {
      const unsigned short* p = base + c * 32 + (lh ? 8 : 0);
      aq[c].u[0] = *reinterpret_cast<const uint4*>(p);
      aq[c].u[1] = *reinterpret_cast<const uint4*>(p + 16);
    }
  }

  f32x8 accO[4];
#pragma unroll
  for (int t = 0; t < 4; ++t) accO[t] = f32x8{};
  float m_run[8], l_run[8];
#pragma unroll
  for (int r = 0; r < 8; ++r) { m_run[r] = -1e30f; l_run[r] = 0.0f; }

  auto stageKV = [&](int buf, int kb) {
    // K block: 32 keys x 64 hd = 256 x 16B chunks
#pragma unroll
    for (int i = 0; i < 2; ++i) {
      int c = tid + i * 128;
      int row = c >> 3, kc = (c & 7) * 8;
      async_copy_b128(lds_off(&sK[buf][row * LDK + kc]),
                      &Kd[(((size_t)bh << 11) + kb + row) * HDIM + kc]);
    }
    // V block: 64 hd x 32 keys = 256 x 16B chunks
#pragma unroll
    for (int i = 0; i < 2; ++i) {
      int c = tid + i * 128;
      int row = c >> 2, sc = (c & 3) * 8;
      async_copy_b128(lds_off(&sV[buf][row * LDV + sc]),
                      &Vt[(((size_t)bh * HDIM + row) << 11) + kb + sc]);
    }
  };

  stageKV(0, 0);
  wait_async0();
  __syncthreads();

  unsigned short* sp = sP[wave];
  int buf = 0;

  for (int kb = 0; kb < SEQ; kb += 32) {
    if (kb + 32 < SEQ) stageKV(buf ^ 1, kb + 32);

    // ---- scores: S[16q x 32k] = Q(16x64) * K^T(64x32)
    f32x8 sc[2]; sc[0] = f32x8{}; sc[1] = f32x8{};
#pragma unroll
    for (int t = 0; t < 2; ++t) {
#pragma unroll
      for (int c = 0; c < 2; ++c) {
        FragAB bk;
        const unsigned short* p = &sK[buf][(t * 16 + lm) * LDK + c * 32 + (lh ? 8 : 0)];
        bk.u[0] = *reinterpret_cast<const uint4*>(p);
        bk.u[1] = *reinterpret_cast<const uint4*>(p + 16);
        sc[t] = __builtin_amdgcn_wmma_f32_16x16x32_bf16(
            false, aq[c].v, false, bk.v, (short)0, sc[t], false, false);
      }
    }

    // ---- online softmax (fp32). C-layout: VGPR r = row, lanes = columns.
    float p0[8], p1[8], alpha[8];
#pragma unroll
    for (int r = 0; r < 8; ++r) {
      float x0 = sc[0][r] * ATT_SCALE;
      float x1 = sc[1][r] * ATT_SCALE;
      float mx = fmaxf(x0, x1);
#pragma unroll
      for (int off = 1; off < 16; off <<= 1)
        mx = fmaxf(mx, __shfl_xor(mx, off, 32));
      float mnew = fmaxf(m_run[r], mx);
      alpha[r] = __expf(m_run[r] - mnew);
      p0[r] = __expf(x0 - mnew);
      p1[r] = __expf(x1 - mnew);
      float rs = p0[r] + p1[r];
#pragma unroll
      for (int off = 1; off < 16; off <<= 1)
        rs += __shfl_xor(rs, off, 32);
      l_run[r] = l_run[r] * alpha[r] + rs;
      m_run[r] = mnew;
    }
#pragma unroll
    for (int t = 0; t < 4; ++t)
#pragma unroll
      for (int r = 0; r < 8; ++r) accO[t][r] *= alpha[r];

    // ---- transpose P (C-layout) -> A-layout via per-wave LDS tile
#pragma unroll
    for (int r = 0; r < 8; ++r) {
      int m = r + 8 * lh;
      sp[m * 40 + lm]      = f2bf(p0[r]);
      sp[m * 40 + 16 + lm] = f2bf(p1[r]);
    }
    __syncthreads();
    FragAB fp;
    {
      const unsigned short* p = sp + lm * 40 + (lh ? 8 : 0);
      fp.u[0] = *reinterpret_cast<const uint4*>(p);
      fp.u[1] = *reinterpret_cast<const uint4*>(p + 16);
    }

    // ---- O += P(16x32) * V(32x64)
#pragma unroll
    for (int t = 0; t < 4; ++t) {
      FragAB bv;
      const unsigned short* p = &sV[buf][(t * 16 + lm) * LDV + (lh ? 8 : 0)];
      bv.u[0] = *reinterpret_cast<const uint4*>(p);
      bv.u[1] = *reinterpret_cast<const uint4*>(p + 16);
      accO[t] = __builtin_amdgcn_wmma_f32_16x16x32_bf16(
          false, fp.v, false, bv.v, (short)0, accO[t], false, false);
    }

    wait_async0();
    __syncthreads();
    buf ^= 1;
  }

  // ---- epilogue: normalize, write bf16 [b, s, h*64+hd]
  const int b = bh >> 4;
  const int h = bh & 15;
#pragma unroll
  for (int r = 0; r < 8; ++r) {
    float inv = 1.0f / l_run[r];
    int row = qb + wave * 16 + r + 8 * lh;
#pragma unroll
    for (int t = 0; t < 4; ++t) {
      int d = h * HDIM + t * 16 + lm;
      O[(((size_t)b << 11) + row) * DMODEL + d] = f2bf(accO[t][r] * inv);
    }
  }
}

// ---------------------------------------------------------------------------
// Launch pipeline
// ---------------------------------------------------------------------------
extern "C" void kernel_launch(void* const* d_in, const int* in_sizes, int n_in,
                              void* d_out, int out_size, void* d_ws, size_t ws_size,
                              hipStream_t stream) {
  const float* x  = (const float*)d_in[0];
  const float* Wq = (const float*)d_in[1];
  const float* Wk = (const float*)d_in[2];
  const float* Wv = (const float*)d_in[3];
  const float* Wo = (const float*)d_in[4];
  float* out = (float*)d_out;

  char* ws = (char*)d_ws;
  const size_t XB = (size_t)MROWS * DMODEL * 2;   // 8 MB bf16 activations
  const size_t WB = (size_t)DMODEL * DMODEL * 2;  // 2 MB bf16 weight
  unsigned short* xbf  = (unsigned short*)(ws);
  unsigned short* wqT  = (unsigned short*)(ws + XB);
  unsigned short* wkT  = (unsigned short*)(ws + XB + 1 * WB);
  unsigned short* wvT  = (unsigned short*)(ws + XB + 2 * WB);
  unsigned short* woT  = (unsigned short*)(ws + XB + 3 * WB);
  unsigned short* qws  = (unsigned short*)(ws + XB + 4 * WB);
  unsigned short* kws  = (unsigned short*)(ws + 2 * XB + 4 * WB);
  unsigned short* vtws = (unsigned short*)(ws + 3 * XB + 4 * WB);
  unsigned short* aws  = (unsigned short*)(ws + 4 * XB + 4 * WB);

  // 0) convert x; transpose+convert all 4 weights
  cvt_f32_bf16<<<2048, 256, 0, stream>>>(x, xbf, MROWS * DMODEL / 4);
  wtrans_bf16<<<dim3(16, 16, 4), 256, 0, stream>>>(Wq, Wk, Wv, Wo, wqT, wkT, wvT, woT);

  // 1-3) projections
  dim3 gGrid(DMODEL / GBN, MROWS / GBM);   // (8, 32)
  gemm_bf16_wmma<<<gGrid, 256, 0, stream>>>(xbf, wqT, qws,  MROWS, DMODEL, DMODEL, 1);
  gemm_bf16_wmma<<<gGrid, 256, 0, stream>>>(xbf, wkT, kws,  MROWS, DMODEL, DMODEL, 1);
  gemm_bf16_wmma<<<gGrid, 256, 0, stream>>>(xbf, wvT, vtws, MROWS, DMODEL, DMODEL, 2);

  // 4) attention
  attn_fwd_wmma<<<dim3(SEQ / 64, BATCH * NHEAD), 128, 0, stream>>>(qws, kws, vtws, aws);

  // 5) output projection -> fp32 d_out
  gemm_bf16_wmma<<<gGrid, 256, 0, stream>>>(aws, woT, out, MROWS, DMODEL, DMODEL, 3);
}
